// Wavenet_39754217292191
// MI455X (gfx1250) — matrix operand order repaired and checked by reading, not compiled, not run on hardware
//
#include <hip/hip_runtime.h>
#include <hip/hip_bf16.h>

// WaveNet on MI455X (gfx1250): all 1x1 convs lowered to bf16 WMMA GEMMs
// (v_wmma_f32_16x16x32_bf16) with M-register-blocking (each weight fragment
// feeds 4 WMMAs), split load/compute phases per k-step so ds_load_b128s
// overlap the matrix pipe, async global->LDS staging, fp32 skip accumulation.

#define BN      16
#define TT      4096
#define INDIM   64
#define LDIM    256
#define ODIM    64
#define NLAYERS 10
#define NPOS    (BN * TT)   // 65536 positions
#define MT64    64          // positions per workgroup tile (layer + gemm)

typedef __attribute__((ext_vector_type(16))) __bf16 bf16x16;
typedef __attribute__((ext_vector_type(8)))  __bf16 bf16x8;
typedef __attribute__((ext_vector_type(8)))  float  f32x8;

union FragU { bf16x16 v; bf16x8 h[2]; };

__device__ __forceinline__ bf16x16 frag_zero() {
  bf16x16 z;
#pragma unroll
  for (int j = 0; j < 16; ++j) z[j] = (__bf16)0.0f;
  return z;
}

// A fragment (16x32 bf16, row-major src, row = lane&15):
// lane half 0 -> K {0..7, 16..23}; half 1 -> K {8..15, 24..31}
__device__ __forceinline__ bf16x16 load_frag_a(const __bf16* p, int ld, int lane) {
  const int r = lane & 15, kh = lane >> 4;
  const __bf16* q = p + (size_t)r * ld + kh * 8;
  FragU u;
  u.h[0] = *(const bf16x8*)(q);
  u.h[1] = *(const bf16x8*)(q + 16);
  return u.v;
}

// B fragment (32x16 bf16): column n = lane&15 is weight row o (W stored [O,K]).
// lane half 0 -> K {0..7, 16..23}; half 1 -> K {8..15, 24..31}
__device__ __forceinline__ bf16x16 load_frag_b(const __bf16* p, int ld, int lane) {
  const int n = lane & 15, kh = lane >> 4;
  const __bf16* q = p + (size_t)n * ld + kh * 8;
  FragU u;
  u.h[0] = *(const bf16x8*)(q);
  u.h[1] = *(const bf16x8*)(q + 16);
  return u.v;
}

__device__ __forceinline__ f32x8 wmma_bf16(bf16x16 a, bf16x16 b, f32x8 c) {
  return __builtin_amdgcn_wmma_f32_16x16x32_bf16(false, a, false, b, (short)0, c,
                                                 false, false);
}

// Async copy of `bytes` (multiple of 16*256) from global to LDS, then drain.
__device__ __forceinline__ void async_copy_to_lds(void* lds_dst, const void* gsrc,
                                                  int bytes, int tid) {
  const unsigned lds0 = (unsigned)(uintptr_t)lds_dst;
  const unsigned long long g0 = (unsigned long long)(uintptr_t)gsrc;
  for (int i = tid; i < bytes / 16; i += 256) {
    unsigned ldsoff = lds0 + (unsigned)(i * 16);
    unsigned long long ga = g0 + (unsigned long long)(i * 16);
    asm volatile("global_load_async_to_lds_b128 %0, %1, off"
                 :: "v"(ldsoff), "v"(ga) : "memory");
  }
  asm volatile("s_wait_asynccnt 0x0" ::: "memory");
}

// ---------------- elementwise prep kernels ----------------
__global__ void convert_kernel(const float* __restrict__ s, __bf16* __restrict__ d, int n) {
  int i = blockIdx.x * 256 + threadIdx.x;
  if (i < n) d[i] = (__bf16)s[i];
}
__global__ void relu_convert_kernel(const float* __restrict__ s, __bf16* __restrict__ d, int n) {
  int i = blockIdx.x * 256 + threadIdx.x;
  if (i < n) d[i] = (__bf16)fmaxf(s[i], 0.0f);
}
// src [*, 2] -> tap1 and tap0 planes
__global__ void split_taps_kernel(const float* __restrict__ s,
                                  __bf16* __restrict__ d1, __bf16* __restrict__ d0, int n) {
  int i = blockIdx.x * 256 + threadIdx.x;
  if (i < n) {
    d1[i] = (__bf16)s[2 * i + 1];
    d0[i] = (__bf16)s[2 * i + 0];
  }
}

// ---------------- generic GEMM: out = act(A[N,K] * W[O,K]^T + bias) ----------------
// 64 rows/block, each wave owns an n-strip and 4 m-tiles (B frag reused 4x).
__global__ __launch_bounds__(256) void gemm_kernel(
    const __bf16* __restrict__ A, const __bf16* __restrict__ W,
    const float* __restrict__ bias, int K, int O, int act,
    __bf16* __restrict__ outB, float* __restrict__ outF) {
  __shared__ __align__(16) __bf16 As[MT64 * 256];
  const int tid = threadIdx.x, wave = tid >> 5, lane = tid & 31;
  const long base = (long)blockIdx.x * MT64;

  {
    const bf16x8* src = (const bf16x8*)(A + base * K);
    for (int i = tid; i < MT64 * K / 8; i += 256) ((bf16x8*)As)[i] = src[i];
  }
  __syncthreads();

  const int n0 = lane & 15, kh = lane >> 4;
  for (int nt = wave; nt < O / 16; nt += 8) {
    f32x8 acc[4];
#pragma unroll
    for (int mt = 0; mt < 4; ++mt)
#pragma unroll
      for (int j = 0; j < 8; ++j) acc[mt][j] = bias[nt * 16 + n0];
    for (int k0 = 0; k0 < K; k0 += 32) {
      bf16x16 b = load_frag_b(W + (size_t)(nt * 16) * K + k0, K, lane);
      bf16x16 a[4];
#pragma unroll
      for (int mt = 0; mt < 4; ++mt) a[mt] = load_frag_a(As + mt * 16 * K + k0, K, lane);
#pragma unroll
      for (int mt = 0; mt < 4; ++mt) acc[mt] = wmma_bf16(a[mt], b, acc[mt]);
    }
#pragma unroll
    for (int mt = 0; mt < 4; ++mt)
#pragma unroll
      for (int j = 0; j < 8; ++j) {
        const int m = mt * 16 + j + 8 * kh;
        const long p = base + m;
        float v = acc[mt][j];
        v = (act == 1) ? fmaxf(v, 0.0f) : ((act == 2) ? tanhf(v) : v);
        if (outB) outB[p * O + nt * 16 + n0] = (__bf16)v;
        if (outF) outF[p * O + nt * 16 + n0] = v;
      }
  }
}

// ---------------- fused WaveNet layer ----------------
// gated = tanh(W1f*h + W0f*h_shift + fb) * sigmoid(W1g*h + W0g*h_shift + gb)
// skip  = Ws*gated + sb ; h_next = skip + Wr*h + rb ; final += skip
__global__ __launch_bounds__(256) void layer_kernel(
    const __bf16* __restrict__ hb, __bf16* __restrict__ hnext, float* __restrict__ finalb,
    const __bf16* __restrict__ f1w, const __bf16* __restrict__ f0w,
    const __bf16* __restrict__ g1w, const __bf16* __restrict__ g0w,
    const __bf16* __restrict__ rww, const __bf16* __restrict__ sww,
    const float* __restrict__ fb, const float* __restrict__ gb,
    const float* __restrict__ rb, const float* __restrict__ sb, int dilate) {
  __shared__ __align__(16) __bf16 hT[MT64 * LDIM];   // 32 KB
  __shared__ __align__(16) __bf16 gT[MT64 * LDIM];   // 32 KB
  const int tid = threadIdx.x, wave = tid >> 5, lane = tid & 31;
  const long base = (long)blockIdx.x * MT64;
  const int tbase = (int)(base & (TT - 1));  // tiles never straddle a batch boundary

  async_copy_to_lds(hT, hb + base * LDIM, MT64 * LDIM * 2, tid);
  __syncthreads();

  const int n0 = lane & 15, kh = lane >> 4;
  const int r = lane & 15;

  // phase 1: gated tile -> LDS (each wave: n-strips nt = wave, wave+8; 4 m-tiles each)
  for (int nt = wave; nt < LDIM / 16; nt += 8) {
    f32x8 aF[4], aG[4];
#pragma unroll
    for (int mt = 0; mt < 4; ++mt)
#pragma unroll
      for (int j = 0; j < 8; ++j) { aF[mt][j] = fb[nt * 16 + n0]; aG[mt][j] = gb[nt * 16 + n0]; }
    // current-time taps (A from LDS)
    for (int k0 = 0; k0 < LDIM; k0 += 32) {
      bf16x16 bF = load_frag_b(f1w + (size_t)(nt * 16) * LDIM + k0, LDIM, lane);
      bf16x16 bG = load_frag_b(g1w + (size_t)(nt * 16) * LDIM + k0, LDIM, lane);
      bf16x16 a[4];
#pragma unroll
      for (int mt = 0; mt < 4; ++mt) a[mt] = load_frag_a(hT + mt * 16 * LDIM + k0, LDIM, lane);
#pragma unroll
      for (int mt = 0; mt < 4; ++mt) {
        aF[mt] = wmma_bf16(a[mt], bF, aF[mt]);
        aG[mt] = wmma_bf16(a[mt], bG, aG[mt]);
      }
    }
    // shifted taps (A from global, per-lane zero predication for t < dilate)
    for (int k0 = 0; k0 < LDIM; k0 += 32) {
      bf16x16 bF = load_frag_b(f0w + (size_t)(nt * 16) * LDIM + k0, LDIM, lane);
      bf16x16 bG = load_frag_b(g0w + (size_t)(nt * 16) * LDIM + k0, LDIM, lane);
      bf16x16 a[4];
#pragma unroll
      for (int mt = 0; mt < 4; ++mt) {
        const bool valid = (tbase + mt * 16 + r >= dilate);
        if (valid) a[mt] = load_frag_a(hb + (base + mt * 16 - (long)dilate) * LDIM + k0, LDIM, lane);
        else       a[mt] = frag_zero();
      }
#pragma unroll
      for (int mt = 0; mt < 4; ++mt) {
        aF[mt] = wmma_bf16(a[mt], bF, aF[mt]);
        aG[mt] = wmma_bf16(a[mt], bG, aG[mt]);
      }
    }
#pragma unroll
    for (int mt = 0; mt < 4; ++mt)
#pragma unroll
      for (int j = 0; j < 8; ++j) {
        const float tv = tanhf(aF[mt][j]);
        const float sv = 1.0f / (1.0f + __expf(-aG[mt][j]));
        const int m = mt * 16 + j + 8 * kh;
        gT[m * LDIM + nt * 16 + n0] = (__bf16)(tv * sv);
      }
  }
  __syncthreads();

  // phase 2: residual (A = hT) + skip (A = gT)
  for (int nt = wave; nt < LDIM / 16; nt += 8) {
    f32x8 aR[4], aS[4];
#pragma unroll
    for (int mt = 0; mt < 4; ++mt)
#pragma unroll
      for (int j = 0; j < 8; ++j) { aR[mt][j] = rb[nt * 16 + n0]; aS[mt][j] = sb[nt * 16 + n0]; }
    for (int k0 = 0; k0 < LDIM; k0 += 32) {
      bf16x16 bR = load_frag_b(rww + (size_t)(nt * 16) * LDIM + k0, LDIM, lane);
      bf16x16 bS = load_frag_b(sww + (size_t)(nt * 16) * LDIM + k0, LDIM, lane);
      bf16x16 a[4], ag[4];
#pragma unroll
      for (int mt = 0; mt < 4; ++mt) a[mt]  = load_frag_a(hT + mt * 16 * LDIM + k0, LDIM, lane);
#pragma unroll
      for (int mt = 0; mt < 4; ++mt) ag[mt] = load_frag_a(gT + mt * 16 * LDIM + k0, LDIM, lane);
#pragma unroll
      for (int mt = 0; mt < 4; ++mt) {
        aR[mt] = wmma_bf16(a[mt],  bR, aR[mt]);
        aS[mt] = wmma_bf16(ag[mt], bS, aS[mt]);
      }
    }
#pragma unroll
    for (int mt = 0; mt < 4; ++mt)
#pragma unroll
      for (int j = 0; j < 8; ++j) {
        const int m = mt * 16 + j + 8 * kh;
        const long p = base + m;
        const float sv = aS[mt][j];
        hnext[p * LDIM + nt * 16 + n0] = (__bf16)(sv + aR[mt][j]);
        finalb[p * LDIM + nt * 16 + n0] += sv;
      }
  }
}

extern "C" void kernel_launch(void* const* d_in, const int* in_sizes, int n_in,
                              void* d_out, int out_size, void* d_ws, size_t ws_size,
                              hipStream_t stream) {
  const float* x      = (const float*)d_in[0];
  const float* emb_w  = (const float*)d_in[1];
  const float* emb_b  = (const float*)d_in[2];
  const float* filt_w = (const float*)d_in[3];
  const float* filt_b = (const float*)d_in[4];
  const float* gate_w = (const float*)d_in[5];
  const float* gate_b = (const float*)d_in[6];
  const float* res_w  = (const float*)d_in[7];
  const float* res_b  = (const float*)d_in[8];
  const float* skip_w = (const float*)d_in[9];
  const float* skip_b = (const float*)d_in[10];
  const float* fin1_w = (const float*)d_in[11];
  const float* fin1_b = (const float*)d_in[12];
  const float* fin2_w = (const float*)d_in[13];
  const float* fin2_b = (const float*)d_in[14];
  const float* out_w  = (const float*)d_in[15];
  const float* out_b  = (const float*)d_in[16];

  const size_t NP = (size_t)NPOS;
  const size_t LL = (size_t)LDIM * LDIM;
  size_t off = 0;
  auto carve = [&](size_t bytes) -> void* {
    void* p = (char*)d_ws + off;
    off = (off + bytes + 255) & ~(size_t)255;
    return p;
  };
  __bf16* xb     = (__bf16*)carve(NP * INDIM * 2);
  __bf16* ping   = (__bf16*)carve(NP * LDIM * 2);
  __bf16* pong   = (__bf16*)carve(NP * LDIM * 2);
  float*  finalb = (float*)carve(NP * LDIM * 4);
  __bf16* embW   = (__bf16*)carve((size_t)LDIM * INDIM * 2);
  __bf16* skipW  = (__bf16*)carve((NLAYERS + 1) * LL * 2);
  __bf16* resW   = (__bf16*)carve(NLAYERS * LL * 2);
  __bf16* f1W    = (__bf16*)carve(NLAYERS * LL * 2);
  __bf16* f0W    = (__bf16*)carve(NLAYERS * LL * 2);
  __bf16* g1W    = (__bf16*)carve(NLAYERS * LL * 2);
  __bf16* g0W    = (__bf16*)carve(NLAYERS * LL * 2);
  __bf16* fin1W  = (__bf16*)carve(LL * 2);
  __bf16* fin2W  = (__bf16*)carve(LL * 2);
  __bf16* outW   = (__bf16*)carve((size_t)ODIM * LDIM * 2);

  dim3 blk(256);
  auto cg = [](int n) { return dim3((unsigned)((n + 255) / 256)); };

  convert_kernel<<<cg(NPOS * INDIM), blk, 0, stream>>>(x, xb, NPOS * INDIM);
  convert_kernel<<<cg(LDIM * INDIM), blk, 0, stream>>>(emb_w, embW, LDIM * INDIM);
  convert_kernel<<<cg((NLAYERS + 1) * (int)LL), blk, 0, stream>>>(skip_w, skipW, (NLAYERS + 1) * (int)LL);
  convert_kernel<<<cg(NLAYERS * (int)LL), blk, 0, stream>>>(res_w, resW, NLAYERS * (int)LL);
  split_taps_kernel<<<cg(NLAYERS * (int)LL), blk, 0, stream>>>(filt_w, f1W, f0W, NLAYERS * (int)LL);
  split_taps_kernel<<<cg(NLAYERS * (int)LL), blk, 0, stream>>>(gate_w, g1W, g0W, NLAYERS * (int)LL);
  convert_kernel<<<cg((int)LL), blk, 0, stream>>>(fin1_w, fin1W, (int)LL);
  convert_kernel<<<cg((int)LL), blk, 0, stream>>>(fin2_w, fin2W, (int)LL);
  convert_kernel<<<cg(ODIM * LDIM), blk, 0, stream>>>(out_w, outW, ODIM * LDIM);

  const dim3 ggrid((unsigned)(NPOS / MT64));
  // embed: h0 = relu(x * embW^T + emb_b)
  gemm_kernel<<<ggrid, blk, 0, stream>>>(xb, embW, emb_b, INDIM, LDIM, 1, ping, nullptr);
  // final = skip_w[0] * h0 + skip_b[0]  (fp32)
  gemm_kernel<<<ggrid, blk, 0, stream>>>(ping, skipW, skip_b, LDIM, LDIM, 0, nullptr, finalb);

  __bf16* hc = ping;
  __bf16* hn = pong;
  for (int i = 0; i < NLAYERS; ++i) {
    layer_kernel<<<ggrid, blk, 0, stream>>>(
        hc, hn, finalb,
        f1W + (size_t)i * LL, f0W + (size_t)i * LL,
        g1W + (size_t)i * LL, g0W + (size_t)i * LL,
        resW + (size_t)i * LL, skipW + (size_t)(i + 1) * LL,
        filt_b + (size_t)i * LDIM, gate_b + (size_t)i * LDIM,
        res_b + (size_t)i * LDIM, skip_b + (size_t)(i + 1) * LDIM, 1 << i);
    __bf16* t = hc; hc = hn; hn = t;
  }

  // final head: relu -> fin1 -> relu -> fin2 -> out_w^T -> tanh
  relu_convert_kernel<<<cg(NPOS * LDIM), blk, 0, stream>>>(finalb, hc, NPOS * LDIM);
  gemm_kernel<<<ggrid, blk, 0, stream>>>(hc, fin1W, fin1_b, LDIM, LDIM, 1, hn, nullptr);
  gemm_kernel<<<ggrid, blk, 0, stream>>>(hn, fin2W, fin2_b, LDIM, LDIM, 0, hc, nullptr);
  gemm_kernel<<<ggrid, blk, 0, stream>>>(hc, outW, out_b, LDIM, ODIM, 2, nullptr, (float*)d_out);
}
